// EAConv_78469052498587
// MI455X (gfx1250) — matrix operand order repaired and checked by reading, not compiled, not run on hardware
//
#include <hip/hip_runtime.h>
#include <hip/hip_bf16.h>

#define DFEAT 128

typedef __attribute__((ext_vector_type(16))) __bf16          v16bf;
typedef __attribute__((ext_vector_type(16))) unsigned short  v16us;
typedef __attribute__((ext_vector_type(8)))  float           v8f;

union FragU {
    uint4 q[2];
    v16us v;
};

static __device__ __forceinline__ unsigned short f32_bf16(float f) {
    unsigned int u = __float_as_uint(f);
    u += 0x7FFFu + ((u >> 16) & 1u);   // round-to-nearest-even
    return (unsigned short)(u >> 16);
}

// ---------------- degree / norm ----------------
__global__ void k_init_deg(float* __restrict__ deg, int n) {
    int i = blockIdx.x * blockDim.x + threadIdx.x;
    if (i < n) deg[i] = 1.0f;                      // self-loop contribution
}

__global__ void k_deg_accum(const int* __restrict__ dst, float* __restrict__ deg, int e) {
    int i = blockIdx.x * blockDim.x + threadIdx.x;
    if (i < e) atomicAdd(&deg[dst[i]], 1.0f);
}

__global__ void k_dinv(float* __restrict__ deg, int n) {
    int i = blockIdx.x * blockDim.x + threadIdx.x;
    if (i < n) deg[i] = rsqrtf(deg[i]);            // deg >= 1 always
}

// ---------------- fp32 -> bf16 ----------------
__global__ void k_to_bf16(const float* __restrict__ in, unsigned short* __restrict__ out, int n) {
    int i = blockIdx.x * blockDim.x + threadIdx.x;
    if (i < n) out[i] = f32_bf16(in[i]);
}

__global__ void k_relu_bf16(const float* __restrict__ in, unsigned short* __restrict__ out, int n) {
    int i = blockIdx.x * blockDim.x + threadIdx.x;
    if (i < n) out[i] = f32_bf16(fmaxf(in[i], 0.0f));
}

// ---------------- bf16 WMMA GEMM: C[M,Nc] = A[M,K] * B[K,Nc] ----------------
#define BM 128
#define BN 64
#define BK 32
#define APAD 8        // row stride 40 shorts = 80 B (16B aligned)
#define BPAD 8

__global__ __launch_bounds__(256)
void k_gemm_bf16(const unsigned short* __restrict__ A,
                 const unsigned short* __restrict__ B,
                 float* __restrict__ C, int M, int K, int Nc) {
    __shared__ unsigned short As [BM][BK + APAD];   // 128 x 40 shorts, row-major
    __shared__ unsigned short Bst[BN][BK + BPAD];   //  64 x 40 shorts, TRANSPOSED (n-major)

    const int tid  = threadIdx.x;
    const int lane = tid & 31;
    const int wave = tid >> 5;                     // 0..7 -> M strip
    const int lh   = lane & 15;                    // N / M sub-index
    const int kh   = lane >> 4;                    // lane-half selector
    const int rowBase = blockIdx.x * BM;
    const int colBase = blockIdx.y * BN;

    v8f acc[4] = {};

    for (int k0 = 0; k0 < K; k0 += BK) {
        // stage A tile (guarded rows), 8-byte chunks, row-major
        for (int c = tid; c < (BM * BK) / 4; c += 256) {
            int r  = c >> 3;                       // 8 chunks per row
            int ch = c & 7;
            int grow = rowBase + r;
            uint2 val = make_uint2(0u, 0u);
            if (grow < M) val = *(const uint2*)(A + (size_t)grow * K + k0 + ch * 4);
            *(uint2*)&As[r][ch * 4] = val;
        }
        // stage B tile transposed: coalesced 8B global read, 4 scalar LDS stores
        for (int c = tid; c < (BK * BN) / 4; c += 256) {
            int r  = c >> 4;                       // k row 0..31
            int ch = c & 15;                       // n chunk
            uint2 val = *(const uint2*)(B + (size_t)(k0 + r) * Nc + colBase + ch * 4);
            const unsigned short* v = (const unsigned short*)&val;
            Bst[ch * 4 + 0][r] = v[0];
            Bst[ch * 4 + 1][r] = v[1];
            Bst[ch * 4 + 2][r] = v[2];
            Bst[ch * 4 + 3][r] = v[3];
        }
        __syncthreads();

        // A fragment: 16-bit A 16x32 layout; per lane two contiguous 16B runs
        //   elems 0..7  -> K = kh*8 .. kh*8+7
        //   elems 8..15 -> K = 16+kh*8 .. 16+kh*8+7
        FragU fa;
        const int am = wave * 16 + lh;
        fa.q[0] = *(const uint4*)&As[am][kh * 8];
        fa.q[1] = *(const uint4*)&As[am][16 + kh * 8];
        v16bf a = __builtin_bit_cast(v16bf, fa.v);

        // fetch ALL four B fragments first (independent ds_load_b128 pairs),
        // then issue the four WMMAs back-to-back behind a single LDS wait
        FragU fb[4];
        #pragma unroll
        for (int j = 0; j < 4; ++j) {
            const unsigned short* bp = &Bst[j * 16 + lh][kh * 8];
            fb[j].q[0] = *(const uint4*)bp;
            fb[j].q[1] = *(const uint4*)(bp + 16);
        }
        #pragma unroll
        for (int j = 0; j < 4; ++j) {
            v16bf b = __builtin_bit_cast(v16bf, fb[j].v);
            acc[j] = __builtin_amdgcn_wmma_f32_16x16x32_bf16(
                false, a, false, b, (short)0, acc[j], false, false);
        }
        __syncthreads();
    }

    // D layout: VGPR e -> M = wave*16 + (lane>=16 ? 8 : 0) + e, N = colBase + j*16 + (lane&15)
    #pragma unroll
    for (int j = 0; j < 4; ++j) {
        int col = colBase + j * 16 + lh;
        #pragma unroll
        for (int e = 0; e < 8; ++e) {
            int row = rowBase + wave * 16 + kh * 8 + e;
            if (row < M) C[(size_t)row * Nc + col] = acc[j][e];
        }
    }
}

// ---------------- out[n,:] = h[n,:]*dinv[n]^2 + bias  (self-loop + bias init) ----------------
__global__ void k_self_bias(const float* __restrict__ h, const float* __restrict__ dinv,
                            const float* __restrict__ bias, float* __restrict__ out,
                            int n, int F4) {
    int i = blockIdx.x * blockDim.x + threadIdx.x;
    if (i >= n * F4) return;
    int node = i / F4;
    int f4   = i % F4;
    float w = dinv[node]; w *= w;
    float4 hv = ((const float4*)h)[(size_t)node * F4 + f4];
    float4 bv = ((const float4*)bias)[f4];
    float4 ov;
    ov.x = hv.x * w + bv.x; ov.y = hv.y * w + bv.y;
    ov.z = hv.z * w + bv.z; ov.w = hv.w * w + bv.w;
    ((float4*)out)[(size_t)node * F4 + f4] = ov;
}

// ---------------- edge scatter: out[dst] += h[src] * dinv[src]*dinv[dst] ----------------
__global__ void k_edge_scatter(const float* __restrict__ h,
                               const int* __restrict__ src, const int* __restrict__ dst,
                               const float* __restrict__ dinv, float* __restrict__ out,
                               int e, int F4) {
    int i = blockIdx.x * blockDim.x + threadIdx.x;
    if (i >= e * F4) return;
    int ed = i / F4;
    int f4 = i % F4;
    int s = src[ed];
    int d = dst[ed];
    float w = dinv[s] * dinv[d];
    float4 hv = ((const float4*)h)[(size_t)s * F4 + f4];
    float* o = out + ((size_t)d * F4 + f4) * 4;
    atomicAdd(o + 0, hv.x * w);
    atomicAdd(o + 1, hv.y * w);
    atomicAdd(o + 2, hv.z * w);
    atomicAdd(o + 3, hv.w * w);
}

// ---------------- launch ----------------
extern "C" void kernel_launch(void* const* d_in, const int* in_sizes, int n_in,
                              void* d_out, int out_size, void* d_ws, size_t ws_size,
                              hipStream_t stream) {
    const float* x  = (const float*)d_in[0];
    const float* W1 = (const float*)d_in[1];
    const float* b1 = (const float*)d_in[2];
    const float* W2 = (const float*)d_in[3];
    const float* b2 = (const float*)d_in[4];
    const int*   ei = (const int*)d_in[5];

    const int N = in_sizes[0] / DFEAT;     // 50000
    const int H = in_sizes[2];             // 512
    const int E = in_sizes[5] / 2;         // 800000
    const int* src = ei;
    const int* dst = ei + E;
    float* out = (float*)d_out;

    // workspace carve (aligned 256B)
    char* ws = (char*)d_ws;
    auto up = [](size_t v) { return (v + 255) & ~(size_t)255; };
    float*          dinv = (float*)ws;          ws += up((size_t)N * 4);
    unsigned short* xb   = (unsigned short*)ws; ws += up((size_t)N * DFEAT * 2);
    unsigned short* W1b  = (unsigned short*)ws; ws += up((size_t)DFEAT * H * 2);
    unsigned short* W2b  = (unsigned short*)ws; ws += up((size_t)H * DFEAT * 2);
    float*          bufA = (float*)ws;          ws += up((size_t)N * H * 4);   // h1, later z1 (bf16)
    float*          bufB = (float*)ws;          ws += up((size_t)N * H * 4);   // out1, later h2
    float*          h1   = bufA;
    unsigned short* z1b  = (unsigned short*)bufA;
    float*          out1 = bufB;
    float*          h2   = bufB;

    const int T = 256;
    auto blk = [&](long long n) { return (unsigned)((n + T - 1) / T); };

    // 1) normalization
    k_init_deg <<<blk(N), T, 0, stream>>>(dinv, N);
    k_deg_accum<<<blk(E), T, 0, stream>>>(dst, dinv, E);
    k_dinv     <<<blk(N), T, 0, stream>>>(dinv, N);

    // 2) bf16 conversions
    k_to_bf16<<<blk((long long)N * DFEAT), T, 0, stream>>>(x,  xb,  N * DFEAT);
    k_to_bf16<<<blk((long long)DFEAT * H), T, 0, stream>>>(W1, W1b, DFEAT * H);
    k_to_bf16<<<blk((long long)H * DFEAT), T, 0, stream>>>(W2, W2b, H * DFEAT);

    // 3) layer 1: h1 = x @ W1  (M=N, K=128, Nc=512)
    {
        dim3 g((N + BM - 1) / BM, H / BN);
        k_gemm_bf16<<<g, 256, 0, stream>>>(xb, W1b, h1, N, DFEAT, H);
    }
    // 4) aggregate: out1 = selfloop + bias, then edge scatter-add
    k_self_bias   <<<blk((long long)N * (H / 4)), T, 0, stream>>>(h1, dinv, b1, out1, N, H / 4);
    k_edge_scatter<<<blk((long long)E * (H / 4)), T, 0, stream>>>(h1, src, dst, dinv, out1, E, H / 4);

    // 5) relu -> bf16 (writes over h1's buffer; reads out1)
    k_relu_bf16<<<blk((long long)N * H), T, 0, stream>>>(out1, z1b, N * H);

    // 6) layer 2: h2 = z1 @ W2  (M=N, K=512, Nc=128) — h2 overwrites out1 (already consumed)
    {
        dim3 g((N + BM - 1) / BM, DFEAT / BN);
        k_gemm_bf16<<<g, 256, 0, stream>>>(z1b, W2b, h2, N, H, DFEAT);
    }
    // 7) final aggregate straight into d_out (init pass fully overwrites poison)
    k_self_bias   <<<blk((long long)N * (DFEAT / 4)), T, 0, stream>>>(h2, dinv, b2, out, N, DFEAT / 4);
    k_edge_scatter<<<blk((long long)E * (DFEAT / 4)), T, 0, stream>>>(h2, src, dst, dinv, out, E, DFEAT / 4);
}
